// GraphTransformer_6794638262758
// MI455X (gfx1250) — matrix operand order repaired and checked by reading, not compiled, not run on hardware
//
#include <hip/hip_runtime.h>
#include <math.h>

// Problem constants (from reference setup_inputs)
#define NB   32        // graphs
#define N0   8192      // initial nodes
#define E0   131072    // edges (fixed length; masked after pooling)
#define EMB  64        // embedding size / per-head dim C
#define HH   8         // heads
#define CC   64        // per-head channels
#define HC   512       // H*C

// GEMM block tiling
#define BM   128
#define BN   64
#define BK   32

typedef __attribute__((ext_vector_type(16))) __bf16 v16bf;
typedef __attribute__((ext_vector_type(8)))  float  v8f;
typedef __attribute__((ext_vector_type(4)))  unsigned int u32x4;
typedef __attribute__((ext_vector_type(8)))  int i32x8;
typedef __attribute__((ext_vector_type(4)))  int i32x4;

// ---------------------------------------------------------------------------
// CAS-based float atomic max
// ---------------------------------------------------------------------------
__device__ __forceinline__ void atomicMaxF(float* addr, float val) {
  int* ai = (int*)addr;
  int old = *ai;
  while (__int_as_float(old) < val) {
    int assumed = old;
    old = atomicCAS(ai, assumed, __float_as_int(val));
    if (old == assumed) break;
  }
}

// ---------------------------------------------------------------------------
// Tensor Data Mover: DMA a dense 2D f32 tile (tile1 rows x tile0 cols) from
// global (row stride = stride0 elements) into LDS at lds_addr (packed dense).
// D# packing per CDNA5 ISA 8.3/8.4 (Group0 128b, Group1 256b; groups 2/3 zero
// for 2D tensors). data_size=2 (4 bytes), type=2 ("image").
// Toolchain probe: this clang exposes the 6-arg builtin
//   (u32x4 g0, i32x8 g1, i32x4 g2, i32x4 g3, i32x8 extra, i32 cpol)
// ---------------------------------------------------------------------------
#if defined(__has_builtin)
#if __has_builtin(__builtin_amdgcn_tensor_load_to_lds)
#define USE_TDM 1
#endif
#endif
#ifndef USE_TDM
#define USE_TDM 0
#endif

#if USE_TDM
__device__ __forceinline__ void tdm_load_2d(unsigned lds_addr, const void* gptr,
                                            unsigned dim0, unsigned dim1,
                                            unsigned stride0,
                                            unsigned tile0, unsigned tile1)
{
  const unsigned long long ga = (unsigned long long)gptr;
  u32x4 g0;
  g0[0] = 1u;                                               // count=1 (valid user D#)
  g0[1] = lds_addr;                                         // LDS byte address
  g0[2] = (unsigned)(ga & 0xffffffffu);                     // global_addr[31:0]
  g0[3] = (unsigned)((ga >> 32) & 0x1ffffffu) | (2u << 30); // addr[56:32] | type=2
  i32x8 g1;
  g1[0] = (int)(2u << 16);                                  // data_size = 4 bytes
  g1[1] = (int)((dim0 & 0xffffu) << 16);                    // tensor_dim0[15:0]
  g1[2] = (int)((dim0 >> 16) | ((dim1 & 0xffffu) << 16));   // dim0[31:16] | dim1[15:0]
  g1[3] = (int)((dim1 >> 16) | ((tile0 & 0xffffu) << 16));  // dim1[31:16] | tile_dim0
  g1[4] = (int)(tile1 & 0xffffu);                           // tile_dim1 (tile_dim2=0)
  g1[5] = (int)stride0;                                     // tensor_dim0_stride[31:0]
  g1[6] = 0;                                                // stride0[47:32], stride1 lo
  g1[7] = 0;                                                // stride1 hi
  const i32x4 z4 = {0, 0, 0, 0};
  const i32x8 z8 = {0, 0, 0, 0, 0, 0, 0, 0};
  __builtin_amdgcn_tensor_load_to_lds(g0, g1, z4, z4, z8, 0);
}
#endif

// ---------------------------------------------------------------------------
// WMMA GEMM: C[M,N] = epilogue(A[M,K] @ W[K,N] + bias)
// fuse: 0 = none, 1 = relu, 2 = relu + batchnorm(eval, eps=1e-5)
// Block = 256 threads (8 waves) computes a BM x BN tile; per K-step wave 0
// issues two TDM descriptors (A slab BMxBK, W slab BKxBN) into LDS, waits on
// TENSORcnt, then each wave computes 16x64 via 4 WMMAs reusing one A fragment.
// Requires: M % 128 == 0, N % 64 == 0, K % 32 == 0 (true for all call sites).
// Register layouts per CDNA5 ISA 7.12.2 (bf16 A 16x32, B 32x16, f32 C 16x16).
// ---------------------------------------------------------------------------
__global__ void __launch_bounds__(256)
gemm_bias_wmma(const float* __restrict__ A, const float* __restrict__ W,
               const float* __restrict__ bias, float* __restrict__ C,
               int M, int K, int N, int fuse,
               const float* __restrict__ gamma, const float* __restrict__ betap,
               const float* __restrict__ mean,  const float* __restrict__ var)
{
  __shared__ float sA[BM][BK];   // 16 KB
  __shared__ float sW[BK][BN];   // 8 KB

  const int wid  = threadIdx.x >> 5;       // wave id: M sub-tile 0..7
  const int lane = threadIdx.x & 31;
  const int nblk = N / BN;
  const int bm = blockIdx.x / nblk;
  const int bn = blockIdx.x % nblk;
  const int l  = lane & 15;
  const int hi = lane >> 4;

  v8f acc[4] = {};

#if USE_TDM
  const unsigned ldsA = (unsigned)(size_t)&sA[0][0];
  const unsigned ldsW = (unsigned)(size_t)&sW[0][0];
#endif

  for (int kb = 0; kb < K; kb += BK) {
    __syncthreads();                       // previous K-step compute done
#if USE_TDM
    if (wid == 0) {                        // one wave drives the TDM (EXEC ignored)
      tdm_load_2d(ldsA, A + (size_t)bm * BM * K + kb, (unsigned)K, (unsigned)M,
                  (unsigned)K, BK, BM);
      tdm_load_2d(ldsW, W + (size_t)kb * N + (size_t)bn * BN, (unsigned)N, (unsigned)K,
                  (unsigned)N, BN, BK);
      __builtin_amdgcn_s_wait_tensorcnt(0);
    }
#else
    for (int idx = threadIdx.x; idx < BM * BK; idx += 256)
      sA[idx / BK][idx % BK] = A[(size_t)(bm * BM + idx / BK) * K + kb + (idx % BK)];
    for (int idx = threadIdx.x; idx < BK * BN; idx += 256)
      sW[idx / BN][idx % BN] = W[(size_t)(kb + idx / BN) * N + bn * BN + (idx % BN)];
#endif
    __syncthreads();

    // A fragment (16x32 bf16): lane row = wid*16 + l;
    // halves e0..7  -> K = hi*8 + e, halves e8..15 -> K = 16 + hi*8 + e
    v16bf a;
    const float* ar = &sA[wid * 16 + l][0];
#pragma unroll
    for (int e = 0; e < 8; ++e) a[e]     = (__bf16)ar[(hi << 3) + e];
#pragma unroll
    for (int e = 0; e < 8; ++e) a[8 + e] = (__bf16)ar[16 + (hi << 3) + e];

#pragma unroll
    for (int t = 0; t < 4; ++t) {
      // B fragment (32x16 bf16): lane col = t*16 + l; halves e0..7 -> K = hi*8 + e, e8..15 -> K = 16 + hi*8 + (e-8)
      v16bf b;
      const int col = t * 16 + l;
#pragma unroll
      for (int e = 0; e < 16; ++e) b[e] = (__bf16)sW[(e < 8 ? (hi << 3) + e : 16 + (hi << 3) + (e - 8))][col];
      acc[t] = __builtin_amdgcn_wmma_f32_16x16x32_bf16(false, a, false, b, (short)0,
                                                       acc[t], false, false);
    }
  }

  // Epilogue: bias (+ relu (+ batchnorm)); C/D layout: VGPR r, lane -> (M=r+8*hi, N=l)
#pragma unroll
  for (int t = 0; t < 4; ++t) {
    const int colg = bn * BN + t * 16 + l;
    const float bc = bias ? bias[colg] : 0.f;
    float g = 1.f, bt = 0.f, mn = 0.f, iv = 1.f;
    if (fuse == 2) {
      g = gamma[colg]; bt = betap[colg]; mn = mean[colg]; iv = rsqrtf(var[colg] + 1e-5f);
    }
#pragma unroll
    for (int r = 0; r < 8; ++r) {
      const int m = bm * BM + wid * 16 + r + (hi << 3);
      float v = acc[t][r] + bc;
      if (fuse >= 1) v = fmaxf(v, 0.f);
      if (fuse == 2) v = (v - mn) * iv * g + bt;
      C[(size_t)m * N + colg] = v;
    }
  }
}

// ---------------------------------------------------------------------------
// Attention edge kernels. Edge term e[h][c] = sum_j edge_attr[e][j]*We[j][h*64+c]
// recomputed on the fly (K=4) rather than materializing [E,512].
// ---------------------------------------------------------------------------
__global__ void edge_logits(const float* __restrict__ q, const float* __restrict__ k,
                            const float* __restrict__ ea, const float* __restrict__ We,
                            const int* __restrict__ esrc, const int* __restrict__ edst,
                            const int* __restrict__ emask,
                            float* __restrict__ logits, float* __restrict__ smax, int E)
{
  const int t = blockIdx.x * blockDim.x + threadIdx.x;
  if (t >= E * HH) return;
  const int eid = t / HH, h = t % HH;
  const int s = esrc[eid], d = edst[eid];
  const float a0 = ea[eid * 4 + 0], a1 = ea[eid * 4 + 1];
  const float a2 = ea[eid * 4 + 2], a3 = ea[eid * 4 + 3];
  const float* qr = q + (size_t)d * HC + h * CC;
  const float* kr = k + (size_t)s * HC + h * CC;
  float dot = 0.f;
  for (int c = 0; c < CC; ++c) {
    const int wc = h * CC + c;
    const float ev = a0 * We[0 * HC + wc] + a1 * We[1 * HC + wc] +
                     a2 * We[2 * HC + wc] + a3 * We[3 * HC + wc];
    dot += qr[c] * (kr[c] + ev);
  }
  const float lg = emask[eid] ? dot * 0.125f : -3.0e38f;  // 1/sqrt(64)
  logits[t] = lg;
  if (emask[eid]) atomicMaxF(&smax[d * HH + h], lg);
}

__global__ void edge_exp(float* __restrict__ elog, const int* __restrict__ edst,
                         const int* __restrict__ emask, const float* __restrict__ smax,
                         float* __restrict__ ssum, int E)
{
  const int t = blockIdx.x * blockDim.x + threadIdx.x;
  if (t >= E * HH) return;
  const int eid = t / HH, h = t % HH;
  const int d = edst[eid];
  float m = smax[d * HH + h];
  m = (m > -1.5e38f) ? m : 0.f;               // empty-segment guard
  const float e = emask[eid] ? __expf(elog[t] - m) : 0.f;
  elog[t] = e;
  if (e != 0.f) atomicAdd(&ssum[d * HH + h], e);
}

__global__ void edge_scatter(const float* __restrict__ elog, const float* __restrict__ ssum,
                             const float* __restrict__ v, const float* __restrict__ ea,
                             const float* __restrict__ We,
                             const int* __restrict__ esrc, const int* __restrict__ edst,
                             const int* __restrict__ emask, float* __restrict__ out, int E)
{
  const int t = blockIdx.x * blockDim.x + threadIdx.x;
  if (t >= E * HH) return;
  const int eid = t / HH, h = t % HH;
  if (!emask[eid]) return;
  const int s = esrc[eid], d = edst[eid];
  const float w = elog[t] / fmaxf(ssum[d * HH + h], 1e-16f);
  if (w == 0.f) return;
  const float a0 = ea[eid * 4 + 0], a1 = ea[eid * 4 + 1];
  const float a2 = ea[eid * 4 + 2], a3 = ea[eid * 4 + 3];
  for (int c = 0; c < CC; ++c) {
    const int wc = h * CC + c;
    const float ev = a0 * We[0 * HC + wc] + a1 * We[1 * HC + wc] +
                     a2 * We[2 * HC + wc] + a3 * We[3 * HC + wc];
    atomicAdd(&out[(size_t)d * HC + wc], w * (v[(size_t)s * HC + wc] + ev));
  }
}

// ---------------------------------------------------------------------------
// Gated skip: beta = sigmoid([out, xr, out-xr] @ Wbeta); x = beta*xr+(1-beta)*out
// ---------------------------------------------------------------------------
__global__ void beta_kernel(const float* __restrict__ out, const float* __restrict__ xr,
                            const float* __restrict__ Wb, float* __restrict__ betav, int N)
{
  const int i = blockIdx.x * blockDim.x + threadIdx.x;
  if (i >= N) return;
  float s = 0.f;
  for (int j = 0; j < HC; ++j) {
    const float o = out[(size_t)i * HC + j], r = xr[(size_t)i * HC + j];
    s += o * Wb[j] + r * Wb[HC + j] + (o - r) * Wb[2 * HC + j];
  }
  betav[i] = 1.f / (1.f + __expf(-s));
}

__global__ void combine_kernel(const float* __restrict__ out, const float* __restrict__ xr,
                               const float* __restrict__ betav, float* __restrict__ x,
                               size_t total)
{
  const size_t t = (size_t)blockIdx.x * blockDim.x + threadIdx.x;
  if (t >= total) return;
  const float b = betav[t / HC];
  x[t] = b * xr[t] + (1.f - b) * out[t];
}

// ---------------------------------------------------------------------------
// TopK pooling: score = tanh(x.w/||w||); rank-select top n_per/2 per graph
// ---------------------------------------------------------------------------
__global__ void score_kernel(const float* __restrict__ x, const float* __restrict__ w,
                             float* __restrict__ score, int N)
{
  const int i = blockIdx.x * blockDim.x + threadIdx.x;
  if (i >= N) return;
  float nn = 0.f;
  for (int j = 0; j < EMB; ++j) nn += w[j] * w[j];
  const float inv = rsqrtf(nn);
  float s = 0.f;
  for (int j = 0; j < EMB; ++j) s += x[(size_t)i * EMB + j] * w[j];
  score[i] = tanhf(s * inv);
}

__global__ void topk_kernel(const float* __restrict__ x, const float* __restrict__ score,
                            float* __restrict__ xnew, int* __restrict__ newid,
                            int n_per, int keep)
{
  __shared__ float sc[256];
  const int g = blockIdx.x;
  const int tIdx = threadIdx.x;               // node within graph; blockDim == n_per
  const int gi = g * n_per + tIdx;
  sc[tIdx] = score[gi];
  __syncthreads();
  const float mys = sc[tIdx];
  int rank = 0;
  for (int j = 0; j < n_per; ++j) {
    const float oj = sc[j];
    if (oj > mys || (oj == mys && j < tIdx)) rank++;   // top_k tie-break: lower index first
  }
  if (rank < keep) {
    const int np = g * keep + rank;
    newid[gi] = np;
    for (int c = 0; c < EMB; ++c)
      xnew[(size_t)np * EMB + c] = x[(size_t)gi * EMB + c] * mys;
  } else {
    newid[gi] = -1;
  }
}

__global__ void edge_remap(const int* __restrict__ newid, int* __restrict__ esrc,
                           int* __restrict__ edst, int* __restrict__ emask, int E)
{
  const int t = blockIdx.x * blockDim.x + threadIdx.x;
  if (t >= E) return;
  const int s2 = newid[esrc[t]], d2 = newid[edst[t]];
  const int valid = emask[t] && s2 >= 0 && d2 >= 0;
  esrc[t] = valid ? s2 : 0;
  edst[t] = valid ? d2 : 0;
  emask[t] = valid;
}

// ---------------------------------------------------------------------------
// Readout: d_out[g][0:64] += max over graph nodes; d_out[g][64:128] += sum/n_per
// ---------------------------------------------------------------------------
__global__ void readout_kernel(const float* __restrict__ x, float* __restrict__ outsum,
                               int n_per)
{
  const int t = blockIdx.x * blockDim.x + threadIdx.x;
  if (t >= NB * EMB) return;
  const int g = t / EMB, c = t % EMB;
  float mx = -3.0e38f, sm = 0.f;
  for (int j = 0; j < n_per; ++j) {
    const float v = x[(size_t)(g * n_per + j) * EMB + c];
    mx = fmaxf(mx, v);
    sm += v;
  }
  outsum[g * 128 + c]      += mx;
  outsum[g * 128 + 64 + c] += sm / (float)n_per;
}

// ---------------------------------------------------------------------------
// Utility fills / copies
// ---------------------------------------------------------------------------
__global__ void fill_f32(float* p, size_t n, float v) {
  const size_t t = (size_t)blockIdx.x * blockDim.x + threadIdx.x;
  if (t < n) p[t] = v;
}
__global__ void fill_i32(int* p, size_t n, int v) {
  const size_t t = (size_t)blockIdx.x * blockDim.x + threadIdx.x;
  if (t < n) p[t] = v;
}
__global__ void copy_i32(int* d, const int* s, size_t n) {
  const size_t t = (size_t)blockIdx.x * blockDim.x + threadIdx.x;
  if (t < n) d[t] = s[t];
}

// ---------------------------------------------------------------------------
// Host orchestration (graph-capture safe: all launches on `stream`, no allocs)
// ---------------------------------------------------------------------------
extern "C" void kernel_launch(void* const* d_in, const int* in_sizes, int n_in,
                              void* d_out, int out_size, void* d_ws, size_t ws_size,
                              hipStream_t stream)
{
  (void)in_sizes; (void)n_in; (void)out_size; (void)ws_size;

  // Input map: insertion-order recursive flatten of setup_inputs()
  //   0:x 1:edge_attr 2:edge_index 3:batch_index
  //   conv params in dict-literal order: Wq,bq,Wk,bk,Wv,bv,We,Wskip,bskip,Wbeta
  const float* X0    = (const float*)d_in[0];
  const float* EATTR = (const float*)d_in[1];
  const int*   EIDX  = (const int*)d_in[2];
  const int IX_CONV1 = 4, IX_TRANSF1 = 14, IX_BN1 = 16,   // bn: gamma,beta,mean,var
            IX_CONVS = 20, IX_TRANSFS = 60, IX_BNS = 68, IX_POOLS = 84;

  // Workspace carve (256B-aligned)
  char* wp = (char*)d_ws;
  auto carve = [&](size_t bytes) { void* p = (void*)wp; wp += (bytes + 255) & ~(size_t)255; return p; };
  float* q     = (float*)carve((size_t)N0 * HC * 4);
  float* kbuf  = (float*)carve((size_t)N0 * HC * 4);
  float* vbuf  = (float*)carve((size_t)N0 * HC * 4);
  float* xr    = (float*)carve((size_t)N0 * HC * 4);
  float* outb  = (float*)carve((size_t)N0 * HC * 4);
  float* xbuf  = (float*)carve((size_t)N0 * HC * 4);
  float* x64a  = (float*)carve((size_t)N0 * EMB * 4);
  float* x64b  = (float*)carve((size_t)N0 * EMB * 4);
  float* elog  = (float*)carve((size_t)E0 * HH * 4);
  float* smax  = (float*)carve((size_t)N0 * HH * 4);
  float* ssum  = (float*)carve((size_t)N0 * HH * 4);
  float* betav = (float*)carve((size_t)N0 * 4);
  float* score = (float*)carve((size_t)N0 * 4);
  int*   newid = (int*)carve((size_t)N0 * 4);
  int*   esrc  = (int*)carve((size_t)E0 * 4);
  int*   edst  = (int*)carve((size_t)E0 * 4);
  int*   emask = (int*)carve((size_t)E0 * 4);

  auto cdiv = [](size_t a, size_t b) { return (unsigned)((a + b - 1) / b); };

  auto gemm = [&](const float* A, const float* W, const float* bias, float* C,
                  int M, int K, int N, int fuse,
                  const float* g, const float* bt, const float* mn, const float* vr) {
    const unsigned blocks = (unsigned)((M / BM) * (N / BN));
    gemm_bias_wmma<<<dim3(blocks), dim3(256), 0, stream>>>(
        A, W, bias, C, M, K, N, fuse, g, bt, mn, vr);
  };

  // Per-call init (deterministic: rebuild edge state + zero output every call)
  copy_i32<<<dim3(cdiv(E0, 256)), dim3(256), 0, stream>>>(esrc, EIDX, (size_t)E0);
  copy_i32<<<dim3(cdiv(E0, 256)), dim3(256), 0, stream>>>(edst, EIDX + E0, (size_t)E0);
  fill_i32<<<dim3(cdiv(E0, 256)), dim3(256), 0, stream>>>(emask, (size_t)E0, 1);
  fill_f32<<<dim3(cdiv(NB * 128, 256)), dim3(256), 0, stream>>>((float*)d_out, (size_t)NB * 128, 0.f);

  auto run_conv = [&](const float* Xin, int N, int K, int base) {
    const float* Wq    = (const float*)d_in[base + 0];
    const float* bq    = (const float*)d_in[base + 1];
    const float* Wk    = (const float*)d_in[base + 2];
    const float* bk    = (const float*)d_in[base + 3];
    const float* Wv    = (const float*)d_in[base + 4];
    const float* bv    = (const float*)d_in[base + 5];
    const float* We    = (const float*)d_in[base + 6];
    const float* Wskip = (const float*)d_in[base + 7];
    const float* bskip = (const float*)d_in[base + 8];
    const float* Wbeta = (const float*)d_in[base + 9];
    gemm(Xin, Wq,    bq,    q,    N, K, HC, 0, nullptr, nullptr, nullptr, nullptr);
    gemm(Xin, Wk,    bk,    kbuf, N, K, HC, 0, nullptr, nullptr, nullptr, nullptr);
    gemm(Xin, Wv,    bv,    vbuf, N, K, HC, 0, nullptr, nullptr, nullptr, nullptr);
    gemm(Xin, Wskip, bskip, xr,   N, K, HC, 0, nullptr, nullptr, nullptr, nullptr);
    fill_f32<<<dim3(cdiv((size_t)N * HH, 256)), dim3(256), 0, stream>>>(smax, (size_t)N * HH, -3.0e38f);
    fill_f32<<<dim3(cdiv((size_t)N * HH, 256)), dim3(256), 0, stream>>>(ssum, (size_t)N * HH, 0.f);
    fill_f32<<<dim3(cdiv((size_t)N * HC, 256)), dim3(256), 0, stream>>>(outb, (size_t)N * HC, 0.f);
    const size_t te = (size_t)E0 * HH;
    edge_logits<<<dim3(cdiv(te, 256)), dim3(256), 0, stream>>>(q, kbuf, EATTR, We, esrc, edst, emask, elog, smax, E0);
    edge_exp<<<dim3(cdiv(te, 256)), dim3(256), 0, stream>>>(elog, edst, emask, smax, ssum, E0);
    edge_scatter<<<dim3(cdiv(te, 256)), dim3(256), 0, stream>>>(elog, ssum, vbuf, EATTR, We, esrc, edst, emask, outb, E0);
    beta_kernel<<<dim3(cdiv((size_t)N, 256)), dim3(256), 0, stream>>>(outb, xr, Wbeta, betav, N);
    combine_kernel<<<dim3(cdiv((size_t)N * HC, 256)), dim3(256), 0, stream>>>(outb, xr, betav, xbuf, (size_t)N * HC);
  };

  // conv1 (din=32) + transf1 + bn1
  run_conv(X0, N0, 32, IX_CONV1);
  gemm(xbuf, (const float*)d_in[IX_TRANSF1], (const float*)d_in[IX_TRANSF1 + 1], x64a,
       N0, HC, EMB, 2,
       (const float*)d_in[IX_BN1 + 0], (const float*)d_in[IX_BN1 + 1],
       (const float*)d_in[IX_BN1 + 2], (const float*)d_in[IX_BN1 + 3]);

  int N = N0, n_per = 256, pool_used = 0;
  float* xcur = x64a;
  float* xalt = x64b;
  for (int i = 0; i < 4; ++i) {
    run_conv(xcur, N, EMB, IX_CONVS + 10 * i);
    gemm(xbuf, (const float*)d_in[IX_TRANSFS + 2 * i], (const float*)d_in[IX_TRANSFS + 2 * i + 1],
         xcur, N, HC, EMB, 2,
         (const float*)d_in[IX_BNS + 4 * i + 0], (const float*)d_in[IX_BNS + 4 * i + 1],
         (const float*)d_in[IX_BNS + 4 * i + 2], (const float*)d_in[IX_BNS + 4 * i + 3]);
    if ((i % 2 == 0) || (i == 3)) {           // top_k_every_n = 2
      const float* pw = (const float*)d_in[IX_POOLS + pool_used];
      pool_used++;
      score_kernel<<<dim3(cdiv((size_t)N, 256)), dim3(256), 0, stream>>>(xcur, pw, score, N);
      const int keep = n_per >> 1;
      topk_kernel<<<dim3(NB), dim3(n_per), 0, stream>>>(xcur, score, xalt, newid, n_per, keep);
      edge_remap<<<dim3(cdiv((size_t)E0, 256)), dim3(256), 0, stream>>>(newid, esrc, edst, emask, E0);
      float* t = xcur; xcur = xalt; xalt = t;
      n_per = keep;
      N = NB * n_per;
    }
    readout_kernel<<<dim3(cdiv((size_t)NB * EMB, 256)), dim3(256), 0, stream>>>(xcur, (float*)d_out, n_per);
  }
}
